// HSTU_34608846471243
// MI455X (gfx1250) — matrix operand order repaired and checked by reading, not compiled, not run on hardware
//
#include <hip/hip_runtime.h>

// ---------------------------------------------------------------------------
// HSTU block for MI455X (gfx1250): bf16 WMMA everywhere GEMM-shaped.
// B=4 T=2048 C_IN=1024 HIDDEN=1024 HEADS=8 HEAD_DIM=128 OUT=1024
// ---------------------------------------------------------------------------

typedef __bf16 bf16;
typedef __attribute__((ext_vector_type(8)))  __bf16 bf16x8;
typedef __attribute__((ext_vector_type(16))) __bf16 bf16x16;
typedef __attribute__((ext_vector_type(8)))  float  f32x8;

#define B_    4
#define T_    2048
#define CIN   1024
#define HID   1024
#define NH    8
#define HD    128
#define ODIM  1024
#define MTOT  (B_ * T_)   // 8192 rows

// ---- fragment helpers (CDNA5 WMMA 16-bit per-lane layouts, ISA 7.12.2) ----

__device__ __forceinline__ bf16x16 cat16(bf16x8 lo, bf16x8 hi) {
  return __builtin_shufflevector(lo, hi, 0,1,2,3,4,5,6,7,8,9,10,11,12,13,14,15);
}

// A (16x32 MxK): lane<16 -> row M=lane, halfs = K{0..7,16..23}; lane>=16 -> K{8..15,24..31}
__device__ __forceinline__ bf16x16 load_a_frag(const bf16* __restrict__ base, int ld, int lane) {
  const bf16* p = base + (size_t)(lane & 15) * ld + ((lane >> 4) << 3);
  return cat16(*(const bf16x8*)p, *(const bf16x8*)(p + 16));
}

// B (32x16 KxN): lane<16 -> col N=lane, halfs = K{0..7,16..23}; lane>=16 -> K{8..15,24..31}
__device__ __forceinline__ bf16x16 load_b_frag(const bf16* __restrict__ base, int ld, int lane) {
  const bf16* p = base + (size_t)(lane & 15) * ld + ((lane >> 4) << 3);
  return cat16(*(const bf16x8*)p, *(const bf16x8*)(p + 16));
}

__device__ __forceinline__ f32x8 wmma_bf16(bf16x16 a, bf16x16 b, f32x8 c) {
  return __builtin_amdgcn_wmma_f32_16x16x32_bf16(false, a, false, b, (short)0, c, false, false);
}

__device__ __forceinline__ float silu_f(float x) { return x / (1.0f + __expf(-x)); }

// ---------------------------------------------------------------------------
// fp32 -> bf16 conversion
// ---------------------------------------------------------------------------
__global__ __launch_bounds__(256) void cvt_kernel(const float* __restrict__ s,
                                                  bf16* __restrict__ d, int n) {
  int i = blockIdx.x * blockDim.x + threadIdx.x;
  if (i < n) d[i] = (bf16)s[i];
}

// ---------------------------------------------------------------------------
// Fused U/V/Q/K projections: out = silu(X @ W^T + b), bf16 outputs.
// blockIdx.z selects the projection; V is stored transposed (B,H,D,T).
// Block = 128 threads = 4 waves; wave tile 32(M) x 64(N); block tile 128x64.
// ---------------------------------------------------------------------------
__global__ __launch_bounds__(128) void proj_kernel(
    const bf16* __restrict__ X,
    const bf16* __restrict__ Wu, const float* __restrict__ bu, bf16* __restrict__ U,
    const bf16* __restrict__ Wv, const float* __restrict__ bv, bf16* __restrict__ Vt,
    const bf16* __restrict__ Wq, const float* __restrict__ bq, bf16* __restrict__ Qo,
    const bf16* __restrict__ Wk, const float* __restrict__ bk, bf16* __restrict__ Ko)
{
  const int lane = threadIdx.x & 31;
  const int wave = threadIdx.x >> 5;
  const int which = blockIdx.z;
  const bf16*  W;
  const float* bias;
  if      (which == 0) { W = Wu; bias = bu; }
  else if (which == 1) { W = Wv; bias = bv; }
  else if (which == 2) { W = Wq; bias = bq; }
  else                 { W = Wk; bias = bk; }
  bf16* Odense = (which == 0) ? U : ((which == 2) ? Qo : Ko);

  const int m0 = blockIdx.x * 128 + wave * 32;
  const int n0 = blockIdx.y * 64;

  f32x8 acc[2][4];
  const f32x8 zero = {0.f,0.f,0.f,0.f,0.f,0.f,0.f,0.f};
  #pragma unroll
  for (int i = 0; i < 2; ++i)
    #pragma unroll
    for (int j = 0; j < 4; ++j) acc[i][j] = zero;

  for (int kk = 0; kk < CIN; kk += 32) {
    if (kk + 64 < CIN) {
      __builtin_prefetch((const void*)(X + (size_t)m0 * CIN + kk + 64), 0, 1);
      __builtin_prefetch((const void*)(W + (size_t)n0 * CIN + kk + 64), 0, 1);
    }
    bf16x16 a0 = load_a_frag(X + (size_t)(m0     ) * CIN + kk, CIN, lane);
    bf16x16 a1 = load_a_frag(X + (size_t)(m0 + 16) * CIN + kk, CIN, lane);
    #pragma unroll
    for (int nf = 0; nf < 4; ++nf) {
      bf16x16 b = load_b_frag(W + (size_t)(n0 + 16 * nf) * CIN + kk, CIN, lane);
      acc[0][nf] = wmma_bf16(a0, b, acc[0][nf]);
      acc[1][nf] = wmma_bf16(a1, b, acc[1][nf]);
    }
  }

  // Epilogue: C/D layout -> VGPR j holds row m0+mi*16+j (+8 for lanes>=16), col n0+(lane&15)
  const int rofs = (lane >> 4) << 3;
  #pragma unroll
  for (int nf = 0; nf < 4; ++nf) {
    const int col = n0 + 16 * nf + (lane & 15);
    const float bsv = bias[col];
    #pragma unroll
    for (int mi = 0; mi < 2; ++mi) {
      #pragma unroll
      for (int j = 0; j < 8; ++j) {
        const int row = m0 + mi * 16 + rofs + j;
        const float v = silu_f(acc[mi][nf][j] + bsv);
        if (which == 1) {
          const int bb = row >> 11, t = row & (T_ - 1);
          const int h  = col >> 7,  d = col & (HD - 1);
          Vt[(((size_t)(bb * NH + h)) * HD + d) * T_ + t] = (bf16)v;
        } else {
          Odense[(size_t)row * HID + col] = (bf16)v;
        }
      }
    }
  }
}

// ---------------------------------------------------------------------------
// Causal HSTU attention: per (b,h) head, per 64-query tile.
// 4 waves x 16 queries; Q fragments resident in registers; per-wave LDS slice
// holds the silu'd score tile as bf16 (A-matrix operand for A@V).
// Normalizer = running row-sum (no softmax max needed).
// ---------------------------------------------------------------------------
__global__ __launch_bounds__(128) void attn_kernel(
    const bf16* __restrict__ Q, const bf16* __restrict__ K,
    const bf16* __restrict__ Vt, float* __restrict__ AV)
{
  __shared__ __align__(16) bf16 sA[4][16][64];

  const int lane = threadIdx.x & 31;
  const int wave = threadIdx.x >> 5;
  const int bh = blockIdx.y;
  const int b  = bh >> 3, h = bh & 7;
  const int m0 = blockIdx.x * 64 + wave * 16;     // first query row (in T)
  const float scale = 0.08838834764831845f;       // 1/sqrt(128)
  const int rofs = (lane >> 4) << 3;

  // Q A-fragments, K dim = 128 -> 4 fragments, held for the whole stream
  bf16x16 qa[4];
  #pragma unroll
  for (int kf = 0; kf < 4; ++kf)
    qa[kf] = load_a_frag(Q + ((size_t)(b * T_) + m0) * HID + h * HD + kf * 32, HID, lane);

  f32x8 av[8];
  const f32x8 zero = {0.f,0.f,0.f,0.f,0.f,0.f,0.f,0.f};
  #pragma unroll
  for (int i = 0; i < 8; ++i) av[i] = zero;
  float rsum = 0.0f;                               // lane L holds row (L&15) sum

  for (int s0 = 0; s0 <= m0 + 15; s0 += 64) {      // causal: stream key tiles
    f32x8 sc[4];
    #pragma unroll
    for (int i = 0; i < 4; ++i) sc[i] = zero;

    // S = Q @ K^T for this 16x64 score tile
    #pragma unroll
    for (int kf = 0; kf < 4; ++kf) {
      #pragma unroll
      for (int nf = 0; nf < 4; ++nf) {
        bf16x16 kb = load_b_frag(
            K + ((size_t)(b * T_) + s0 + 16 * nf) * HID + h * HD + kf * 32, HID, lane);
        sc[nf] = wmma_bf16(qa[kf], kb, sc[nf]);
      }
    }

    // scale + silu + relu + causal mask, stash bf16 A-tile in this wave's LDS slice
    #pragma unroll
    for (int nf = 0; nf < 4; ++nf) {
      const int scol = s0 + 16 * nf + (lane & 15);
      #pragma unroll
      for (int j = 0; j < 8; ++j) {
        const int row = m0 + rofs + j;
        const float x = sc[nf][j] * scale;
        const float a = (x > 0.0f && scol <= row) ? (x / (1.0f + __expf(-x))) : 0.0f;
        sA[wave][rofs + j][16 * nf + (lane & 15)] = (bf16)a;
      }
    }

    // running row sums (same-wave LDS ops are in order; no barrier needed)
    {
      const bf16* rp = &sA[wave][lane & 15][0];
      float rs = 0.0f;
      #pragma unroll
      for (int c = 0; c < 64; c += 8) {
        bf16x8 vv = *(const bf16x8*)(rp + c);
        #pragma unroll
        for (int e = 0; e < 8; ++e) rs += (float)vv[e];
      }
      rsum += rs;
    }

    // AV += A @ V   (A from LDS, V pre-transposed so B-fragments are contiguous)
    #pragma unroll
    for (int kt = 0; kt < 2; ++kt) {
      bf16x16 af = load_a_frag(&sA[wave][0][kt * 32], 64, lane);
      #pragma unroll
      for (int nf = 0; nf < 8; ++nf) {
        bf16x16 vb = load_b_frag(
            Vt + (((size_t)(b * NH + h)) * HD + 16 * nf) * T_ + s0 + kt * 32, T_, lane);
        av[nf] = wmma_bf16(af, vb, av[nf]);
      }
    }
  }

  // normalize by row-sum and store fp32 AV
  #pragma unroll
  for (int nf = 0; nf < 8; ++nf) {
    const int d = 16 * nf + (lane & 15);
    #pragma unroll
    for (int j = 0; j < 8; ++j) {
      const int row = m0 + rofs + j;
      const float rs = __shfl(rsum, rofs + j, 32);
      const float s  = (rs > 1e-12f) ? (1.0f / (rs + 1e-8f)) : 0.0f;
      AV[((size_t)(b * T_) + row) * HID + h * HD + d] = av[nf][j] * s;
    }
  }
}

// ---------------------------------------------------------------------------
// RMSNorm(AV) * g * U  -> bf16 Y.  One 256-thread block per (b,t) row.
// ---------------------------------------------------------------------------
__global__ __launch_bounds__(256) void rmsnorm_mul_kernel(
    const float* __restrict__ AV, const bf16* __restrict__ U,
    const float* __restrict__ g, bf16* __restrict__ Y)
{
  __shared__ float red[8];
  const int row = blockIdx.x;
  const float* av = AV + (size_t)row * HID;

  float v[4];
  float ss = 0.0f;
  #pragma unroll
  for (int i = 0; i < 4; ++i) {
    v[i] = av[threadIdx.x + 256 * i];
    ss += v[i] * v[i];
  }
  #pragma unroll
  for (int off = 16; off > 0; off >>= 1) ss += __shfl_xor(ss, off, 32);
  if ((threadIdx.x & 31) == 0) red[threadIdx.x >> 5] = ss;
  __syncthreads();
  float tot = 0.0f;
  #pragma unroll
  for (int i = 0; i < 8; ++i) tot += red[i];

  const float inv = rsqrtf(tot * (1.0f / (float)HID) + 1.1920929e-07f);
  #pragma unroll
  for (int i = 0; i < 4; ++i) {
    const int c = threadIdx.x + 256 * i;
    const float y = v[i] * inv * g[c] * (float)U[(size_t)row * HID + c];
    Y[(size_t)row * HID + c] = (bf16)y;
  }
}

// ---------------------------------------------------------------------------
// Final projection: Out = Y @ Wf^T + bf, fp32 output.
// ---------------------------------------------------------------------------
__global__ __launch_bounds__(128) void final_gemm_kernel(
    const bf16* __restrict__ Y, const bf16* __restrict__ Wf,
    const float* __restrict__ bias, float* __restrict__ Out)
{
  const int lane = threadIdx.x & 31;
  const int wave = threadIdx.x >> 5;
  const int m0 = blockIdx.x * 128 + wave * 32;
  const int n0 = blockIdx.y * 64;

  f32x8 acc[2][4];
  const f32x8 zero = {0.f,0.f,0.f,0.f,0.f,0.f,0.f,0.f};
  #pragma unroll
  for (int i = 0; i < 2; ++i)
    #pragma unroll
    for (int j = 0; j < 4; ++j) acc[i][j] = zero;

  for (int kk = 0; kk < HID; kk += 32) {
    if (kk + 64 < HID) {
      __builtin_prefetch((const void*)(Y  + (size_t)m0 * HID + kk + 64), 0, 1);
      __builtin_prefetch((const void*)(Wf + (size_t)n0 * HID + kk + 64), 0, 1);
    }
    bf16x16 a0 = load_a_frag(Y + (size_t)(m0     ) * HID + kk, HID, lane);
    bf16x16 a1 = load_a_frag(Y + (size_t)(m0 + 16) * HID + kk, HID, lane);
    #pragma unroll
    for (int nf = 0; nf < 4; ++nf) {
      bf16x16 b = load_b_frag(Wf + (size_t)(n0 + 16 * nf) * HID + kk, HID, lane);
      acc[0][nf] = wmma_bf16(a0, b, acc[0][nf]);
      acc[1][nf] = wmma_bf16(a1, b, acc[1][nf]);
    }
  }

  const int rofs = (lane >> 4) << 3;
  #pragma unroll
  for (int nf = 0; nf < 4; ++nf) {
    const int col = n0 + 16 * nf + (lane & 15);
    const float bsv = bias[col];
    #pragma unroll
    for (int mi = 0; mi < 2; ++mi) {
      #pragma unroll
      for (int j = 0; j < 8; ++j) {
        const int row = m0 + mi * 16 + rofs + j;
        Out[(size_t)row * ODIM + col] = acc[mi][nf][j] + bsv;
      }
    }
  }
}

// ---------------------------------------------------------------------------
// Host launcher
// ---------------------------------------------------------------------------
extern "C" void kernel_launch(void* const* d_in, const int* in_sizes, int n_in,
                              void* d_out, int out_size, void* d_ws, size_t ws_size,
                              hipStream_t stream)
{
  (void)in_sizes; (void)n_in; (void)out_size; (void)ws_size;

  const float* x   = (const float*)d_in[0];
  // d_in[1] = attn_mask (causal tril) -- implicit in the kernel
  const float* Wu  = (const float*)d_in[2];
  const float* bu  = (const float*)d_in[3];
  const float* Wv  = (const float*)d_in[4];
  const float* bv  = (const float*)d_in[5];
  const float* Wq  = (const float*)d_in[6];
  const float* bq  = (const float*)d_in[7];
  const float* Wk  = (const float*)d_in[8];
  const float* bk  = (const float*)d_in[9];
  const float* Wf  = (const float*)d_in[10];
  const float* bfb = (const float*)d_in[11];
  const float* g   = (const float*)d_in[12];
  float* out = (float*)d_out;

  // workspace carve-out
  char* ws = (char*)d_ws;
  size_t off = 0;
  auto alloc = [&](size_t bytes) -> char* {
    char* p = ws + off;
    off += (bytes + 255) & ~(size_t)255;
    return p;
  };
  bf16* X16  = (bf16*)alloc((size_t)MTOT * CIN * sizeof(bf16));
  bf16* Wu16 = (bf16*)alloc((size_t)HID * CIN * sizeof(bf16));
  bf16* Wv16 = (bf16*)alloc((size_t)HID * CIN * sizeof(bf16));
  bf16* Wq16 = (bf16*)alloc((size_t)HID * CIN * sizeof(bf16));
  bf16* Wk16 = (bf16*)alloc((size_t)HID * CIN * sizeof(bf16));
  bf16* Wf16 = (bf16*)alloc((size_t)ODIM * HID * sizeof(bf16));
  bf16* U16  = (bf16*)alloc((size_t)MTOT * HID * sizeof(bf16));
  bf16* Q16  = (bf16*)alloc((size_t)MTOT * HID * sizeof(bf16));
  bf16* K16  = (bf16*)alloc((size_t)MTOT * HID * sizeof(bf16));
  bf16* Vt16 = (bf16*)alloc((size_t)MTOT * HID * sizeof(bf16));   // (B,H,D,T)
  float* AV32 = (float*)alloc((size_t)MTOT * HID * sizeof(float));
  bf16* Y16  = (bf16*)alloc((size_t)MTOT * HID * sizeof(bf16));

  // 1) fp32 -> bf16 conversions
  const int nX = MTOT * CIN;
  const int nW = HID * CIN;
  cvt_kernel<<<(nX + 255) / 256, 256, 0, stream>>>(x,  X16,  nX);
  cvt_kernel<<<(nW + 255) / 256, 256, 0, stream>>>(Wu, Wu16, nW);
  cvt_kernel<<<(nW + 255) / 256, 256, 0, stream>>>(Wv, Wv16, nW);
  cvt_kernel<<<(nW + 255) / 256, 256, 0, stream>>>(Wq, Wq16, nW);
  cvt_kernel<<<(nW + 255) / 256, 256, 0, stream>>>(Wk, Wk16, nW);
  cvt_kernel<<<(nW + 255) / 256, 256, 0, stream>>>(Wf, Wf16, nW);

  // 2) fused U/V/Q/K projections (silu), V transposed
  dim3 gp(MTOT / 128, HID / 64, 4);
  proj_kernel<<<gp, 128, 0, stream>>>(X16,
                                      Wu16, bu, U16,
                                      Wv16, bv, Vt16,
                                      Wq16, bq, Q16,
                                      Wk16, bk, K16);

  // 3) causal attention with row-sum normalization
  dim3 ga(T_ / 64, B_ * NH);
  attn_kernel<<<ga, 128, 0, stream>>>(Q16, K16, Vt16, AV32);

  // 4) RMSNorm * g * U -> Y (bf16)
  rmsnorm_mul_kernel<<<MTOT, 256, 0, stream>>>(AV32, U16, g, Y16);

  // 5) final projection -> fp32 output
  dim3 gf(MTOT / 128, ODIM / 64);
  final_gemm_kernel<<<gf, 128, 0, stream>>>(Y16, Wf16, bfb, out);
}
